// DualNet_SS_22230750724212
// MI455X (gfx1250) — matrix operand order repaired and checked by reading, not compiled, not run on hardware
//
#include <hip/hip_runtime.h>
#include <hip/hip_bf16.h>

typedef __bf16 bf16;
typedef __attribute__((ext_vector_type(16))) __bf16 v16bf;
typedef __attribute__((ext_vector_type(8)))  __bf16 bf16x8;
typedef __attribute__((ext_vector_type(4)))  __bf16 bf16x4;
typedef __attribute__((ext_vector_type(8)))  float   v8f;
typedef __attribute__((ext_vector_type(4)))  float   f32x4;

#define CAT16(lo,hi) __builtin_shufflevector(lo,hi,0,1,2,3,4,5,6,7,8,9,10,11,12,13,14,15)

// gfx1250 async global->LDS copy path (ASYNCcnt). Guarded so compile can never
// regress if the toolchain lacks the builtins; fallback = VGPR-staged ds_store.
#if defined(__AMDGCN__) && \
    __has_builtin(__builtin_amdgcn_global_load_async_to_lds_b128) && \
    __has_builtin(__builtin_amdgcn_s_wait_asynccnt)
#define USE_ASYNC_LDS 1
#else
#define USE_ASYNC_LDS 0
#endif

#if USE_ASYNC_LDS
// builtin signature (from hipcc diagnostic): (v4i AS1*, v4i AS3*, Imm int, Imm int)
typedef int v4i __attribute__((__vector_size__(4 * sizeof(int))));
typedef __attribute__((address_space(1))) v4i* gas_b128;
typedef __attribute__((address_space(3))) v4i* las_b128;
#endif

static constexpr int   BATCH = 65536;
static constexpr int   DIM   = 256;
static constexpr float EPSV  = 1e-5f;
static constexpr float NEGS  = 0.01f;

static constexpr int BM = 128, BN = 256, BK = 32;
static constexpr int PADK = BK + 8;          // LDS row stride in halves (80B, 16B aligned)

__device__ __forceinline__ float leaky(float v) { return v >= 0.f ? v : NEGS * v; }

// ---------------------------------------------------------------------------
// Tiled WMMA GEMM: C[M=65536, N=256] = A[M,K] * W[N,K]^T  (bf16 MACs, f32 acc)
// ASRC: 0 = A from f32 with per-row flag mask (stage 1)
//       1 = A from bf16 workspace (stages 2,3) -> async copy eligible
//       2 = A concat: K<256 from f32 sfix (no mask), K>=256 from f32 x*flag
// EPI : 0 = y*scale[n]+bias[n], leaky -> bf16 out (folded BatchNorm)
//       1 = raw f32 out
//       2 = GroupNorm(8-ch groups via shfl_xor) + leaky, out = sfix + flag*y
// ---------------------------------------------------------------------------
template<int ASRC, int EPI>
__launch_bounds__(256)
__global__ void gemm256(const float* __restrict__ Af, const bf16* __restrict__ Ah,
                        const float* __restrict__ A2f, const int* __restrict__ flags,
                        const bf16* __restrict__ W1, const bf16* __restrict__ W2,
                        const float* __restrict__ e0, const float* __restrict__ e1,
                        const float* __restrict__ sfixp,
                        float* __restrict__ outf, bf16* __restrict__ outh, int K)
{
  __shared__ bf16 Al[2][BM][PADK];
  __shared__ bf16 Bl[2][BN][PADK];

  const int tid  = threadIdx.x;
  const int lane = tid & 31;
  const int wid  = tid >> 5;       // 8 waves
  const int wm   = wid >> 1;       // 0..3  (32-row strip)
  const int wn   = wid & 1;        // 0..1  (128-col strip)
  const int rl   = lane & 15;
  const int hs   = lane >> 4;
  const int rowBase = blockIdx.x * BM;
  const int nk = K / BK;

  f32x4  pfA[4];
  bf16x8 pfAh[2];
  bf16x8 pfB[4];
  v16bf  afrag[2];
  v8f    acc[2][8];
  #pragma unroll
  for (int mt = 0; mt < 2; ++mt)
    #pragma unroll
    for (int nt = 0; nt < 8; ++nt)
      #pragma unroll
      for (int j = 0; j < 8; ++j) acc[mt][nt][j] = 0.f;

  // ---- A tile: global f32/bf16 -> registers (with mask / conversion) ----
  auto prefetchA = [&](int kb) {
    const int k0 = kb * BK;
    if constexpr (ASRC == 0) {
      #pragma unroll
      for (int it = 0; it < 4; ++it) {
        int flat = it * 256 + tid, r = flat >> 3, c = flat & 7;
        int grow = rowBase + r;
        f32x4 v = *(const f32x4*)(Af + (size_t)grow * DIM + k0 + c * 4);
        float fl = (float)flags[grow];
        #pragma unroll
        for (int j = 0; j < 4; ++j) v[j] *= fl;
        pfA[it] = v;
      }
    } else if constexpr (ASRC == 2) {
      #pragma unroll
      for (int it = 0; it < 4; ++it) {
        int flat = it * 256 + tid, r = flat >> 3, c = flat & 7;
        int grow = rowBase + r;
        f32x4 v;
        if (k0 < 256) {
          v = *(const f32x4*)(Af + (size_t)grow * DIM + k0 + c * 4);
        } else {
          v = *(const f32x4*)(A2f + (size_t)grow * DIM + (k0 - 256) + c * 4);
          float fl = (float)flags[grow];
          #pragma unroll
          for (int j = 0; j < 4; ++j) v[j] *= fl;
        }
        pfA[it] = v;
      }
    } else {  // ASRC == 1: bf16 source (only used on fallback path)
      #pragma unroll
      for (int it = 0; it < 2; ++it) {
        int flat = it * 256 + tid, r = flat >> 2, c = flat & 3;
        pfAh[it] = *(const bf16x8*)(Ah + (size_t)(rowBase + r) * DIM + k0 + c * 8);
      }
    }
  };

  auto commitA = [&](int bsel) {
    if constexpr (ASRC == 1) {
      #pragma unroll
      for (int it = 0; it < 2; ++it) {
        int flat = it * 256 + tid, r = flat >> 2, c = flat & 3;
        *(bf16x8*)&Al[bsel][r][c * 8] = pfAh[it];
      }
    } else {
      #pragma unroll
      for (int it = 0; it < 4; ++it) {
        int flat = it * 256 + tid, r = flat >> 3, c = flat & 7;
        bf16x4 h;
        #pragma unroll
        for (int j = 0; j < 4; ++j) h[j] = (__bf16)pfA[it][j];
        *(bf16x4*)&Al[bsel][r][c * 4] = h;
      }
    }
  };

  // ---- B tile: bf16 weights -> registers (fallback path) ----
  auto prefetchB = [&](int kb) {
    const int k0 = kb * BK;
    const bf16* Wp = W1; int kk = k0;
    if constexpr (ASRC == 2) { if (k0 >= 256) { Wp = W2; kk = k0 - 256; } }
    #pragma unroll
    for (int it = 0; it < 4; ++it) {
      int flat = it * 256 + tid, n = flat >> 2, c = flat & 3;
      pfB[it] = *(const bf16x8*)(Wp + (size_t)n * DIM + kk + c * 8);
    }
  };

  auto commitB = [&](int bsel) {
    #pragma unroll
    for (int it = 0; it < 4; ++it) {
      int flat = it * 256 + tid, n = flat >> 2, c = flat & 3;
      *(bf16x8*)&Bl[bsel][n][c * 8] = pfB[it];
    }
  };

#if USE_ASYNC_LDS
  // ---- async global->LDS copies (no VGPR staging, tracked by ASYNCcnt) ----
  auto asyncB = [&](int kb, int bsel) {
    const int k0 = kb * BK;
    const bf16* Wp = W1; int kk = k0;
    if constexpr (ASRC == 2) { if (k0 >= 256) { Wp = W2; kk = k0 - 256; } }
    #pragma unroll
    for (int it = 0; it < 4; ++it) {
      int flat = it * 256 + tid, n = flat >> 2, c = flat & 3;
      __builtin_amdgcn_global_load_async_to_lds_b128(
          (gas_b128)(Wp + (size_t)n * DIM + kk + c * 8),
          (las_b128)&Bl[bsel][n][c * 8], 0, 0);
    }
  };
  auto asyncA = [&](int kb, int bsel) {   // ASRC == 1 only (bf16 source)
    const int k0 = kb * BK;
    #pragma unroll
    for (int it = 0; it < 2; ++it) {
      int flat = it * 256 + tid, r = flat >> 2, c = flat & 3;
      __builtin_amdgcn_global_load_async_to_lds_b128(
          (gas_b128)(Ah + (size_t)(rowBase + r) * DIM + k0 + c * 8),
          (las_b128)&Al[bsel][r][c * 8], 0, 0);
    }
  };
#endif

  auto compute = [&](int bsel) {
    #pragma unroll
    for (int mt = 0; mt < 2; ++mt) {       // A frag: M=lane&15, K blocks of 8 interleaved
      int r = wm * 32 + mt * 16 + rl;
      const bf16* p = &Al[bsel][r][hs * 8];
      bf16x8 lo = *(const bf16x8*)p;
      bf16x8 hi = *(const bf16x8*)(p + 16);
      afrag[mt] = CAT16(lo, hi);
    }
    #pragma unroll
    for (int nt = 0; nt < 8; ++nt) {       // B frag: N=lane&15, K blocks of 8 interleaved
      int n = wn * 128 + nt * 16 + rl;
      const bf16* p = &Bl[bsel][n][hs * 8];
      bf16x8 lo = *(const bf16x8*)p;
      bf16x8 hi = *(const bf16x8*)(p + 16);
      v16bf bfrag = CAT16(lo, hi);
      acc[0][nt] = __builtin_amdgcn_wmma_f32_16x16x32_bf16(
          false, afrag[0], false, bfrag, (short)0, acc[0][nt], false, false);
      acc[1][nt] = __builtin_amdgcn_wmma_f32_16x16x32_bf16(
          false, afrag[1], false, bfrag, (short)0, acc[1][nt], false, false);
    }
  };

  // ------------------------- main loop (double-buffered) -------------------------
#if USE_ASYNC_LDS
  if constexpr (ASRC == 1) { asyncA(0, 0); } else { prefetchA(0); commitA(0); }
  asyncB(0, 0);
  __builtin_amdgcn_s_wait_asynccnt(0);
  __syncthreads();
  for (int kb = 0; kb < nk; ++kb) {
    int cur = kb & 1;
    bool more = (kb + 1 < nk);
    if (more) {                       // target buffer drained before last barrier
      if constexpr (ASRC == 1) asyncA(kb + 1, cur ^ 1);
      else                     prefetchA(kb + 1);
      asyncB(kb + 1, cur ^ 1);
    }
    compute(cur);
    if (more) {
      if constexpr (ASRC != 1) commitA(cur ^ 1);
      __builtin_amdgcn_s_wait_asynccnt(0);
      __syncthreads();
    }
  }
#else
  prefetchA(0); prefetchB(0);
  commitA(0);   commitB(0);
  __syncthreads();
  for (int kb = 0; kb < nk; ++kb) {
    int cur = kb & 1;
    bool more = (kb + 1 < nk);
    if (more) { prefetchA(kb + 1); prefetchB(kb + 1); }
    compute(cur);
    if (more) { commitA(cur ^ 1); commitB(cur ^ 1); __syncthreads(); }
  }
#endif

  // ------------------------- epilogue -------------------------
  #pragma unroll
  for (int mt = 0; mt < 2; ++mt) {
    #pragma unroll
    for (int nt = 0; nt < 8; ++nt) {
      #pragma unroll
      for (int i = 0; i < 8; ++i) {
        int grow = rowBase + wm * 32 + mt * 16 + i + hs * 8;   // C: VGPR i -> M=i / M=i+8
        int gcol = wn * 128 + nt * 16 + rl;                    //     lanes -> N
        float v = acc[mt][nt][i];
        size_t idx = (size_t)grow * DIM + gcol;
        if constexpr (EPI == 0) {
          v = leaky(v * e0[gcol] + e1[gcol]);
          outh[idx] = (bf16)v;
        } else if constexpr (EPI == 1) {
          outf[idx] = v;
        } else {
          // GroupNorm: 8-channel group == 8 adjacent lanes of this VGPR
          float s = v, q = v * v;
          s += __shfl_xor(s, 1); q += __shfl_xor(q, 1);
          s += __shfl_xor(s, 2); q += __shfl_xor(q, 2);
          s += __shfl_xor(s, 4); q += __shfl_xor(q, 4);
          float mu  = s * 0.125f;
          float var = q * 0.125f - mu * mu;
          float y   = (v - mu) * rsqrtf(var + EPSV);
          y = leaky(y * e0[gcol] + e1[gcol]);
          float sv = sfixp[idx];
          float fl = (float)flags[grow];
          outf[idx] = sv + fl * y;        // flag==1: compos ; flag==0: sfix
        }
      }
    }
  }
}

// ---------------------------------------------------------------------------
// small prep / glue kernels
// ---------------------------------------------------------------------------
__global__ void cvt_bf16_k(const float* __restrict__ s, bf16* __restrict__ d, int n) {
  int i = blockIdx.x * blockDim.x + threadIdx.x;
  if (i < n) d[i] = (bf16)s[i];
}

__global__ void split_wc_k(const float* __restrict__ Wc, bf16* __restrict__ W1,
                           bf16* __restrict__ W2) {  // Wc: [256, 512]
  int i = blockIdx.x * blockDim.x + threadIdx.x;     // 131072
  int n = i >> 9, k = i & 511;
  if (k < 256) W1[n * 256 + k] = (bf16)Wc[i];
  else         W2[n * 256 + (k - 256)] = (bf16)Wc[i];
}

__global__ void bn_prep_k(const float* g1, const float* b1, const float* m1, const float* v1,
                          const float* g2, const float* b2, const float* m2, const float* v2,
                          float* s1, float* t1, float* s2, float* t2) {
  int i = threadIdx.x;  // 256
  float a = g1[i] * rsqrtf(v1[i] + EPSV); s1[i] = a; t1[i] = b1[i] - m1[i] * a;
  float c = g2[i] * rsqrtf(v2[i] + EPSV); s2[i] = c; t2[i] = b2[i] - m2[i] * c;
}

__global__ void fixup_k(float* __restrict__ sT, float* __restrict__ sC, float* __restrict__ sL,
                        const int* __restrict__ fT, const int* __restrict__ fC,
                        const int* __restrict__ fL) {
  int b = blockIdx.x, n = threadIdx.x;
  float ft = (float)fT[b], fc = (float)fC[b], fl = (float)fL[b];
  float cnt = ft + fc + fl;
  size_t idx = (size_t)b * DIM + n;
  float vT = sT[idx], vC = sC[idx], vL = sL[idx];
  float mean = (ft * vT + fc * vC + fl * vL) / fmaxf(cnt, 1.f);
  bool has = cnt > 0.f;
  sT[idx] = (ft == 0.f && has) ? mean : vT;
  sC[idx] = (fc == 0.f && has) ? mean : vC;
  sL[idx] = (fl == 0.f && has) ? mean : vL;
}

__global__ void logits_k(const float* __restrict__ fuT, const float* __restrict__ fuC,
                         const float* __restrict__ fuL, const float* __restrict__ Wcls,
                         const float* __restrict__ bcls, float* __restrict__ logits) {
  __shared__ float red[256];
  int b = blockIdx.x, n = threadIdx.x;
  size_t idx = (size_t)b * DIM + n;
  float fa = (fuT[idx] + fuC[idx] + fuL[idx]) * (1.f / 3.f);
  red[n] = fa * Wcls[n];
  __syncthreads();
  for (int s = 128; s > 0; s >>= 1) {
    if (n < s) red[n] += red[n + s];
    __syncthreads();
  }
  if (n == 0) logits[b] = red[0] + bcls[0];
}

// ---------------------------------------------------------------------------
extern "C" void kernel_launch(void* const* d_in, const int* in_sizes, int n_in,
                              void* d_out, int out_size, void* d_ws, size_t ws_size,
                              hipStream_t stream) {
  const float* X[3]  = {(const float*)d_in[0], (const float*)d_in[1], (const float*)d_in[2]};
  const int*   F[3]  = {(const int*)d_in[3],   (const int*)d_in[4],   (const int*)d_in[5]};
  const float* Wr    = (const float*)d_in[6];
  const float* g1    = (const float*)d_in[7];
  const float* b1    = (const float*)d_in[8];
  const float* m1    = (const float*)d_in[9];
  const float* v1    = (const float*)d_in[10];
  const float* Wa    = (const float*)d_in[11];
  const float* g2    = (const float*)d_in[12];
  const float* b2    = (const float*)d_in[13];
  const float* m2    = (const float*)d_in[14];
  const float* v2    = (const float*)d_in[15];
  const float* Wo    = (const float*)d_in[16];
  const float* Wc    = (const float*)d_in[17];
  const float* gng   = (const float*)d_in[18];
  const float* gnb   = (const float*)d_in[19];
  const float* Wcls  = (const float*)d_in[20];
  const float* bcls  = (const float*)d_in[21];

  // workspace layout
  char* ws = (char*)d_ws;
  size_t off = 0;
  auto take = [&](size_t bytes) { void* p = ws + off; off = (off + bytes + 255) & ~(size_t)255; return p; };
  bf16* Wr_h  = (bf16*)take(65536 * 2);
  bf16* Wa_h  = (bf16*)take(65536 * 2);
  bf16* Wo_h  = (bf16*)take(65536 * 2);
  bf16* Wc1_h = (bf16*)take(65536 * 2);
  bf16* Wc2_h = (bf16*)take(65536 * 2);
  float* bn1s = (float*)take(256 * 4);
  float* bn1t = (float*)take(256 * 4);
  float* bn2s = (float*)take(256 * 4);
  float* bn2t = (float*)take(256 * 4);
  const size_t modBytesH = (size_t)BATCH * DIM * 2;
  bf16* H1[3], *H2[3];
  for (int m = 0; m < 3; ++m) H1[m] = (bf16*)take(modBytesH);
  for (int m = 0; m < 3; ++m) H2[m] = (bf16*)take(modBytesH);

  // output layout: logits[B] | sT | sC | sL | fusedT | fusedC | fusedL
  float* out = (float*)d_out;
  float* logits = out;
  const size_t BD = (size_t)BATCH * DIM;
  float* S[3]  = {out + BATCH,          out + BATCH + BD,     out + BATCH + 2 * BD};
  float* FU[3] = {out + BATCH + 3 * BD, out + BATCH + 4 * BD, out + BATCH + 5 * BD};

  // --- prep ---
  cvt_bf16_k<<<256, 256, 0, stream>>>(Wr, Wr_h, 65536);
  cvt_bf16_k<<<256, 256, 0, stream>>>(Wa, Wa_h, 65536);
  cvt_bf16_k<<<256, 256, 0, stream>>>(Wo, Wo_h, 65536);
  split_wc_k<<<512, 256, 0, stream>>>(Wc, Wc1_h, Wc2_h);
  bn_prep_k<<<1, 256, 0, stream>>>(g1, b1, m1, v1, g2, b2, m2, v2, bn1s, bn1t, bn2s, bn2t);

  const dim3 gg(BATCH / BM), gb(256);
  // --- stage 1: h1 = leaky(bn1(x_masked @ Wr^T)) ---
  for (int m = 0; m < 3; ++m)
    gemm256<0, 0><<<gg, gb, 0, stream>>>(X[m], nullptr, nullptr, F[m], Wr_h, nullptr,
                                         bn1s, bn1t, nullptr, nullptr, H1[m], 256);
  // --- stage 2: h2 = leaky(bn2(h1 @ Wa^T)) ---
  for (int m = 0; m < 3; ++m)
    gemm256<1, 0><<<gg, gb, 0, stream>>>(nullptr, H1[m], nullptr, nullptr, Wa_h, nullptr,
                                         bn2s, bn2t, nullptr, nullptr, H2[m], 256);
  // --- stage 3: s = h2 @ Wo^T  (raw, into s*_fix slots) ---
  for (int m = 0; m < 3; ++m)
    gemm256<1, 1><<<gg, gb, 0, stream>>>(nullptr, H2[m], nullptr, nullptr, Wo_h, nullptr,
                                         nullptr, nullptr, nullptr, S[m], nullptr, 256);
  // --- stage 4: cross-modality mean / where fix (in place) ---
  fixup_k<<<BATCH, 256, 0, stream>>>(S[0], S[1], S[2], F[0], F[1], F[2]);
  // --- stage 5: fused = sfix + flag * leaky(GN([sfix|x_m] @ Wc^T)) ---
  for (int m = 0; m < 3; ++m)
    gemm256<2, 2><<<gg, gb, 0, stream>>>(S[m], nullptr, X[m], F[m], Wc1_h, Wc2_h,
                                         gng, gnb, S[m], FU[m], nullptr, 512);
  // --- stage 6: logits ---
  logits_k<<<BATCH, 256, 0, stream>>>(FU[0], FU[1], FU[2], Wcls, bcls, logits);
}
